// KNNAttention_30623116820628
// MI455X (gfx1250) — matrix operand order repaired and checked by reading, not compiled, not run on hardware
//
#include <hip/hip_runtime.h>
#include <math.h>

#define BATCH   2
#define DIMC    512
#define NHEADS  8
#define DHEAD   64
#define NTOK    4096          // 64*64 spatial positions
#define KMAX    32
#define KMIN    4
#define NEG_INF (-3.402823466e+38f)

typedef __attribute__((ext_vector_type(16))) __bf16 v16bf;
typedef __attribute__((ext_vector_type(8)))  float  v8f;

// ---------------------------------------------------------------------------
// Kernel 0a: f32 -> bf16 conversion (weights: already K-contiguous row-major)
// ---------------------------------------------------------------------------
__global__ __launch_bounds__(256) void cvt_f32_bf16(const float* __restrict__ src,
                                                    __bf16* __restrict__ dst, int n) {
#if defined(__HIP_DEVICE_COMPILE__)
  int i = blockIdx.x * blockDim.x + threadIdx.x;
  int stride = gridDim.x * blockDim.x;
  for (; i < n; i += stride) dst[i] = (__bf16)src[i];
#endif
}

// ---------------------------------------------------------------------------
// Kernel 0b: LDS-tiled transpose  x f32 [b, c, n]  ->  xt bf16 [b, n, c]
// Makes the qkv GEMM B-operand (fixed n, contiguous c) a pure b128 stream.
// grid (NTOK/32, DIMC/32, B), block 256
// ---------------------------------------------------------------------------
__global__ __launch_bounds__(256) void transpose_x(const float* __restrict__ x,
                                                   __bf16* __restrict__ xt) {
#if defined(__HIP_DEVICE_COMPILE__)
  __shared__ float tile[32][33];
  const int b  = blockIdx.z;
  const int n0 = blockIdx.x * 32;
  const int c0 = blockIdx.y * 32;
  const int tx = threadIdx.x & 31;
  const int ty = threadIdx.x >> 5;                 // 0..7
  const float* xp = x + (size_t)b * DIMC * NTOK;
  #pragma unroll
  for (int r = 0; r < 4; ++r)                      // coalesced over n
    tile[ty + 8 * r][tx] = xp[(size_t)(c0 + ty + 8 * r) * NTOK + n0 + tx];
  __syncthreads();
  __bf16* xo = xt + (size_t)b * NTOK * DIMC;
  #pragma unroll
  for (int r = 0; r < 4; ++r)                      // coalesced over c
    xo[(size_t)(n0 + ty + 8 * r) * DIMC + c0 + tx] = (__bf16)tile[tx][ty + 8 * r];
#endif
}

// ---------------------------------------------------------------------------
// Kernel 1: adaptive-tau gate -> per-query dynamic k  (one block per (b,n))
// ---------------------------------------------------------------------------
__global__ __launch_bounds__(128) void gate_kernel(const float* __restrict__ x,
                                                   const float* __restrict__ w1,
                                                   const float* __restrict__ b1,
                                                   const float* __restrict__ w2,
                                                   const float* __restrict__ b2,
                                                   int* __restrict__ kdyn) {
#if defined(__HIP_DEVICE_COMPILE__)
  const int bn = blockIdx.x;
  const int b  = bn / NTOK;
  const int n  = bn % NTOK;
  const int t  = threadIdx.x;

  const float* xp = x + (size_t)b * DIMC * NTOK + n;
  const float* wp = w1 + (size_t)t * DIMC;

  float acc = b1[t];
  #pragma unroll 8
  for (int c = 0; c < DIMC; ++c) acc += wp[c] * xp[(size_t)c * NTOK];
  acc = acc > 0.f ? acc : 0.f;
  acc *= w2[t];

  __shared__ float red[128];
  red[t] = acc;
  __syncthreads();
  for (int s = 64; s > 0; s >>= 1) {
    if (t < s) red[t] += red[t + s];
    __syncthreads();
  }
  if (t == 0) {
    float g   = red[0] + b2[0];
    float tau = 1.f / (1.f + __expf(-g));
    int   kd  = (int)(tau * (float)KMAX);
    kd = kd < KMIN ? KMIN : (kd > KMAX ? KMAX : kd);
    kdyn[bn] = kd;
  }
#endif
}

// ---------------------------------------------------------------------------
// Kernel 2: qkv GEMM  Y[o,n] = sum_c W[o,c] * Xt[b,n,c]   (bf16 WMMA, f32 acc)
// Both operands contiguous per lane. Packed b128 epilogue into q/k/v [b,h,n,d].
// grid (NTOK/64, 3C/16, B), block 128 (4 waves x one 16x16 tile)
// ---------------------------------------------------------------------------
__global__ __launch_bounds__(128) void gemm_qkv(const __bf16* __restrict__ W,
                                                const __bf16* __restrict__ Xt,
                                                __bf16* __restrict__ qo,
                                                __bf16* __restrict__ ko,
                                                float*  __restrict__ vo) {
#if defined(__HIP_DEVICE_COMPILE__)
  const int b      = blockIdx.z;
  const int m0     = blockIdx.y * 16;
  const int wave   = threadIdx.x >> 5;
  const int n0     = blockIdx.x * 64 + wave * 16;
  const int lane   = threadIdx.x & 31;
  const int laneLo = lane & 15;
  const int sel    = lane >> 4;

  const __bf16* Arow = W + (size_t)(m0 + laneLo) * DIMC + sel * 16;
  const __bf16* Brow = Xt + ((size_t)b * NTOK + n0 + laneLo) * DIMC + sel * 16;

  v8f c = {};
  for (int k0 = 0; k0 < DIMC; k0 += 32) {
    v16bf a, bb;
    #pragma unroll
    for (int e = 0; e < 8; ++e) {               // A: K = k0+sel*16+e (contiguous)
      a[e]     = Arow[k0 + e];
      a[8 + e] = Arow[k0 + 8 + e];
    }
    #pragma unroll
    for (int e = 0; e < 16; ++e)                // B: K = k0+sel*16+e (contiguous!)
      bb[e] = Brow[k0 + e];
    c = __builtin_amdgcn_wmma_f32_16x16x32_bf16(false, a, false, bb,
                                                (short)0, c, false, false);
  }

  const int ncol = n0 + laneLo;
  const int o0   = m0 + sel * 8;                // 8 consecutive output channels
  if (o0 < DIMC) {                              // ---- q (fold 1/sqrt(64))
    const int h = o0 >> 6, d0 = o0 & 63;
    __bf16 tmp[8];
    #pragma unroll
    for (int r = 0; r < 8; ++r) tmp[r] = (__bf16)(c[r] * 0.125f);
    *(uint4*)(qo + (((size_t)(b * NHEADS + h) * NTOK + ncol) * DHEAD + d0)) =
        *(const uint4*)tmp;
  } else if (o0 < 2 * DIMC) {                   // ---- k
    const int oo = o0 - DIMC, h = oo >> 6, d0 = oo & 63;
    __bf16 tmp[8];
    #pragma unroll
    for (int r = 0; r < 8; ++r) tmp[r] = (__bf16)c[r];
    *(uint4*)(ko + (((size_t)(b * NHEADS + h) * NTOK + ncol) * DHEAD + d0)) =
        *(const uint4*)tmp;
  } else {                                      // ---- v stays f32
    const int oo = o0 - 2 * DIMC, h = oo >> 6, d0 = oo & 63;
    float* vp = vo + (((size_t)(b * NHEADS + h) * NTOK + ncol) * DHEAD + d0);
    float4 lo = make_float4(c[0], c[1], c[2], c[3]);
    float4 hi = make_float4(c[4], c[5], c[6], c[7]);
    *(float4*)vp       = lo;
    *(float4*)(vp + 4) = hi;
  }
#endif
}

// ---------------------------------------------------------------------------
// Kernel 3: fused sim (WMMA) -> running top-32 -> dynamic-k softmax -> attn*V
// Key tiles staged once per block via global_load_async_to_lds_b128 (double
// buffered, ASYNCcnt-pipelined); all 4 waves build B-fragments from LDS.
// grid (NTOK/64, NHEADS, B), block 128 (4 waves x 16 queries)
// ---------------------------------------------------------------------------
__global__ __launch_bounds__(128) void attn_kernel(const __bf16* __restrict__ q,
                                                   const __bf16* __restrict__ k,
                                                   const float*  __restrict__ v,
                                                   const int*    __restrict__ kdyn,
                                                   __bf16* __restrict__ att) {
#if defined(__HIP_DEVICE_COMPILE__)
  const int b      = blockIdx.z;
  const int h      = blockIdx.y;
  const int wave   = threadIdx.x >> 5;
  const int i0     = blockIdx.x * 64 + wave * 16;
  const int lane   = threadIdx.x & 31;
  const int laneLo = lane & 15;
  const int sel    = lane >> 4;

  __shared__ __align__(16) __bf16 ktile[2][16][DHEAD];   // 2 x 2KB key tiles
  __shared__ float topv[4][16][33];
  __shared__ int   topi[4][16][33];
  __shared__ float sc[4][16][17];

  const size_t bh     = (size_t)(b * NHEADS + h);
  const __bf16* qrow  = q + (bh * NTOK + (i0 + laneLo)) * DHEAD;
  const __bf16* kbase = k + bh * NTOK * DHEAD;
  const float*  vbase = v + bh * NTOK * DHEAD;

  // A fragments: 16-query tile, K(=dh) 0..31 and 32..63 (contiguous loads)
  v16bf a0, a1;
  #pragma unroll
  for (int e = 0; e < 8; ++e) {
    a0[e]     = qrow[sel * 16 + e];
    a0[8 + e] = qrow[sel * 16 + 8 + e];
    a1[e]     = qrow[32 + sel * 16 + e];
    a1[8 + e] = qrow[32 + sel * 16 + 8 + e];
  }

  float minv = NEG_INF;
  int   minslot = 0;
  if (sel == 0) {
    for (int s = 0; s < KMAX; ++s) { topv[wave][lane][s] = NEG_INF; topi[wave][lane][s] = 0; }
  }

  // async staging: thread -> (row, 16-byte segment) of the 16x64 bf16 key tile
  const int srow = threadIdx.x >> 3;            // 0..15
  const int sseg = (threadIdx.x & 7) * 8;       // 0,8,...,56 (bf16 elements)
  {
    unsigned l0 = (unsigned)(size_t)&ktile[0][srow][sseg];
    const __bf16* g0 = kbase + (size_t)srow * DHEAD + sseg;
    asm volatile("global_load_async_to_lds_b128 %0, %1, off"
                 :: "v"(l0), "v"(g0) : "memory");
  }

  int buf = 0;
  for (int j0 = 0; j0 < NTOK; j0 += 16) {
    if (j0 + 16 < NTOK) {                       // issue next tile, wait current
      unsigned l1 = (unsigned)(size_t)&ktile[buf ^ 1][srow][sseg];
      const __bf16* g1 = kbase + (size_t)(j0 + 16 + srow) * DHEAD + sseg;
      asm volatile("global_load_async_to_lds_b128 %0, %1, off"
                   :: "v"(l1), "v"(g1) : "memory");
      asm volatile("s_wait_asynccnt 0x1" ::: "memory");
    } else {
      asm volatile("s_wait_asynccnt 0x0" ::: "memory");
    }
    __syncthreads();                            // tile ready; prev sc readers done

    // B fragments straight from LDS (ds_load_b128-class reads)
    const __bf16* krow = &ktile[buf][laneLo][sel * 16];
    v16bf b0, b1;
    #pragma unroll
    for (int e = 0; e < 16; ++e) {
      b0[e] = krow[e];
      b1[e] = krow[32 + e];
    }
    v8f cacc = {};
    cacc = __builtin_amdgcn_wmma_f32_16x16x32_bf16(false, a0, false, b0,
                                                   (short)0, cacc, false, false);
    cacc = __builtin_amdgcn_wmma_f32_16x16x32_bf16(false, a1, false, b1,
                                                   (short)0, cacc, false, false);

    #pragma unroll
    for (int r = 0; r < 8; ++r) sc[wave][r + sel * 8][laneLo] = cacc[r];
    __syncthreads();                            // scores visible

    if (sel == 0) {                             // lane owns query row `lane`
      for (int cc = 0; cc < 16; ++cc) {
        float val = sc[wave][lane][cc];
        if (val > minv) {
          topv[wave][lane][minslot] = val;
          topi[wave][lane][minslot] = j0 + cc;
          float mv = topv[wave][lane][0];
          int   ms = 0;
          for (int s = 1; s < KMAX; ++s) {
            float tv = topv[wave][lane][s];
            if (tv < mv) { mv = tv; ms = s; }
          }
          minv = mv; minslot = ms;
        }
      }
    }
    buf ^= 1;
  }

  if (sel == 0) {
    // sort top-32 descending
    for (int s = 1; s < KMAX; ++s) {
      float kv = topv[wave][lane][s];
      int   ki = topi[wave][lane][s];
      int   t2 = s - 1;
      while (t2 >= 0 && topv[wave][lane][t2] < kv) {
        topv[wave][lane][t2 + 1] = topv[wave][lane][t2];
        topi[wave][lane][t2 + 1] = topi[wave][lane][t2];
        --t2;
      }
      topv[wave][lane][t2 + 1] = kv;
      topi[wave][lane][t2 + 1] = ki;
    }
    // dynamic-k mask + softmax; topv becomes weights
    const int kd = kdyn[b * NTOK + i0 + lane];
    const float m = topv[wave][lane][0];
    float sum = 0.f;
    for (int s = 0; s < KMAX; ++s) {
      float wv = (s < kd) ? __expf(topv[wave][lane][s] - m) : 0.f;
      sum += wv;
      topv[wave][lane][s] = wv;
    }
    const float inv = 1.f / sum;
    for (int s = 0; s < KMAX; ++s) topv[wave][lane][s] *= inv;
  }
  __syncthreads();

  // weighted V gather: lane owns 2 of dh=64 dims; packed coalesced att store
  const int d0 = lane * 2;
  unsigned* attu = (unsigned*)att;
  for (int r = 0; r < 16; ++r) {
    const int i = i0 + r;
    float acc0 = 0.f, acc1 = 0.f;
    for (int s = 0; s < KMAX; ++s) {
      const float wgt = topv[wave][r][s];
      const int   idx = topi[wave][r][s];
      const float* vrow = vbase + (size_t)idx * DHEAD;
      acc0 += wgt * vrow[d0];
      acc1 += wgt * vrow[d0 + 1];
    }
    union { __bf16 hh[2]; unsigned u; } pk;
    pk.hh[0] = (__bf16)acc0;
    pk.hh[1] = (__bf16)acc1;
    // att_t[b, n, c] with c = h*64 + d : contiguous per lane
    attu[(((size_t)b * NTOK + i) * DIMC + h * DHEAD) / 2 + lane] = pk.u;
  }
#endif
}

// ---------------------------------------------------------------------------
// Kernel 4: proj GEMM + bias + residual; B-operand = att_t[b,n,c] contiguous
// grid (NTOK/64, C/16, B), block 128
// ---------------------------------------------------------------------------
__global__ __launch_bounds__(128) void gemm_proj(const __bf16* __restrict__ Wp,
                                                 const __bf16* __restrict__ At,
                                                 const float* __restrict__ xin,
                                                 const float* __restrict__ bproj,
                                                 float* __restrict__ out) {
#if defined(__HIP_DEVICE_COMPILE__)
  const int b      = blockIdx.z;
  const int m0     = blockIdx.y * 16;
  const int wave   = threadIdx.x >> 5;
  const int n0     = blockIdx.x * 64 + wave * 16;
  const int lane   = threadIdx.x & 31;
  const int laneLo = lane & 15;
  const int sel    = lane >> 4;

  const __bf16* Arow = Wp + (size_t)(m0 + laneLo) * DIMC + sel * 16;
  const __bf16* Brow = At + ((size_t)b * NTOK + n0 + laneLo) * DIMC + sel * 16;

  v8f c = {};
  for (int k0 = 0; k0 < DIMC; k0 += 32) {
    v16bf a, bb;
    #pragma unroll
    for (int e = 0; e < 8; ++e) {
      a[e]     = Arow[k0 + e];
      a[8 + e] = Arow[k0 + 8 + e];
    }
    #pragma unroll
    for (int e = 0; e < 16; ++e) bb[e] = Brow[k0 + e];
    c = __builtin_amdgcn_wmma_f32_16x16x32_bf16(false, a, false, bb,
                                                (short)0, c, false, false);
  }

  const int ncol = n0 + laneLo;
  #pragma unroll
  for (int r = 0; r < 8; ++r) {
    const int o = m0 + r + sel * 8;
    const size_t idx = ((size_t)b * DIMC + o) * NTOK + ncol;
    out[idx] = xin[idx] + bproj[o] + c[r];
  }
#endif
}

// ---------------------------------------------------------------------------
extern "C" void kernel_launch(void* const* d_in, const int* in_sizes, int n_in,
                              void* d_out, int out_size, void* d_ws, size_t ws_size,
                              hipStream_t stream) {
  const float* x      = (const float*)d_in[0];
  const float* w_qkv  = (const float*)d_in[1];
  const float* w_proj = (const float*)d_in[2];
  const float* b_proj = (const float*)d_in[3];
  const float* w_g1   = (const float*)d_in[4];
  const float* b_g1   = (const float*)d_in[5];
  const float* w_g2   = (const float*)d_in[6];
  const float* b_g2   = (const float*)d_in[7];
  float* out = (float*)d_out;

  char* w = (char*)d_ws;
  size_t off = 0;
  auto carve = [&](size_t bytes) -> void* {
    void* p = w + off;
    off = (off + bytes + 255) & ~(size_t)255;
    return p;
  };
  const size_t nX = (size_t)BATCH * DIMC * NTOK;
  __bf16* xt     = (__bf16*)carve(nX * 2);                  // x^T bf16 [b,n,c]
  __bf16* wqkvb  = (__bf16*)carve((size_t)3 * DIMC * DIMC * 2);
  __bf16* wprojb = (__bf16*)carve((size_t)DIMC * DIMC * 2);
  __bf16* qb     = (__bf16*)carve(nX * 2);                  // [b,h,n,d]
  __bf16* kb     = (__bf16*)carve(nX * 2);                  // [b,h,n,d]
  float*  vf     = (float*)carve(nX * 4);                   // [b,h,n,d]
  __bf16* att_t  = (__bf16*)carve(nX * 2);                  // [b,n,c]
  int*    kdyn   = (int*)carve((size_t)BATCH * NTOK * 4);
  (void)ws_size; (void)in_sizes; (void)n_in; (void)out_size;

  // 0) conversions / transpose
  transpose_x<<<dim3(NTOK / 32, DIMC / 32, BATCH), 256, 0, stream>>>(x, xt);
  cvt_f32_bf16<<<1024, 256, 0, stream>>>(w_qkv, wqkvb, 3 * DIMC * DIMC);
  cvt_f32_bf16<<<512, 256, 0, stream>>>(w_proj, wprojb, DIMC * DIMC);

  // 1) gate -> dynamic k
  gate_kernel<<<BATCH * NTOK, 128, 0, stream>>>(x, w_g1, b_g1, w_g2, b_g2, kdyn);

  // 2) qkv GEMM (WMMA bf16)
  gemm_qkv<<<dim3(NTOK / 64, 3 * DIMC / 16, BATCH), 128, 0, stream>>>(
      wqkvb, xt, qb, kb, vf);

  // 3) fused sim + top-k + softmax + gather (async LDS key staging)
  attn_kernel<<<dim3(NTOK / 64, NHEADS, BATCH), 128, 0, stream>>>(
      qb, kb, vf, kdyn, att_t);

  // 4) proj GEMM + bias + residual
  gemm_proj<<<dim3(NTOK / 64, DIMC / 16, BATCH), 128, 0, stream>>>(
      wprojb, att_t, x, b_proj, out);
}
